// FlashInfer_QKNorm_317827580192
// MI455X (gfx1250) — matrix operand order repaired and checked by reading, not compiled, not run on hardware
//
#include <hip/hip_runtime.h>
#include <stdint.h>

#define MM 1024
#define NN 4096
#define DD 128
#define PP 3072
#define HH 32
#define KVLEN (PP + MM)

typedef __bf16 bf16;
typedef __attribute__((ext_vector_type(16))) __bf16 v16bf;
typedef __attribute__((ext_vector_type(8)))  float  v8f;

union FragBF { v16bf v; uint32_t u[8]; __bf16 h[16]; };
union Acc    { v8f v; float f[8]; };

__device__ __forceinline__ bf16 us2bf(unsigned short v) {
  union { unsigned short s; __bf16 b; } u; u.s = v; return u.b;
}

// ---- WMMA fragment loaders (CDNA5 16-bit layouts, wave32) ----
// A 16x32 (MxK): lane half h, row = lane&15; dword i holds K = ((i>>2)<<4) + 8h + 2(i&3)
__device__ __forceinline__ v16bf frag_a_rowmajor(const bf16* base, int stride, int lane) {
  const int half = lane >> 4;
  const bf16* row = base + (size_t)(lane & 15) * stride;
  FragBF f;
#pragma unroll
  for (int i = 0; i < 8; ++i) {
    int k = ((i >> 2) << 4) + half * 8 + (i & 3) * 2;
    f.u[i] = *reinterpret_cast<const uint32_t*>(row + k);
  }
  return f.v;
}

// B 32x16 (KxN) from LDS holding B^T (bt[n][k], k pairs contiguous):
// col = lane&15; dword i holds K = ((i>>2)<<4) + 8*half + 2(i&3)
__device__ __forceinline__ v16bf frag_b_transposed(const bf16* bt, int stride, int lane) {
  const int half = lane >> 4;
  const bf16* col = bt + (size_t)(lane & 15) * stride;
  FragBF f;
#pragma unroll
  for (int i = 0; i < 8; ++i) {
    int k = ((i >> 2) << 4) + half * 8 + (i & 3) * 2;
    f.u[i] = *reinterpret_cast<const uint32_t*>(col + k);
  }
  return f.v;
}

// ---------------- QKV projection GEMM (+fused per-head RMS norm) ----------------
#define BM 128
#define BN 128
#define BK 32
#define AST 36
#define BST 36

__global__ __launch_bounds__(256)
void qkv_proj_kernel(const float* __restrict__ X, const float* __restrict__ W,
                     bf16* __restrict__ out, int do_norm)
{
  __shared__ bf16  Asb[BM * AST];      // X tile [m][k], bf16
  __shared__ bf16  Bsb[BN * BST];      // W tile transposed [n][k], bf16
  __shared__ float rowstat[BM];

  const int t    = threadIdx.x;
  const int lane = t & 31;
  const int wave = t >> 5;
  const int wm = wave & 3;             // rows: wm*32 + {0,16}
  const int wn = wave >> 2;            // cols: wn*64 + {0,16,32,48}
  const int m0 = blockIdx.y * BM;
  const int n0 = blockIdx.x * BN;
  const int head = blockIdx.x;         // BN == DD -> one head per block column

  const float4* Xv = reinterpret_cast<const float4*>(X);
  const float4* Wv = reinterpret_cast<const float4*>(W);

  Acc acc[2][4];
#pragma unroll
  for (int i = 0; i < 2; ++i)
#pragma unroll
    for (int j = 0; j < 4; ++j)
#pragma unroll
      for (int e = 0; e < 8; ++e) acc[i][j].f[e] = 0.f;

  for (int kt = 0; kt < NN / BK; ++kt) {
    const int kbase = kt * BK;
    // stage A: 128x32 f32 -> bf16, b128 loads, contiguous in k
#pragma unroll
    for (int i = 0; i < 4; ++i) {
      int q4 = t + i * 256;
      int r = q4 >> 3, c4 = q4 & 7;          // 8 float4 per row
      float4 x = Xv[(size_t)(m0 + r) * (NN / 4) + (kbase >> 2) + c4];
      int o = r * AST + c4 * 4;
      Asb[o + 0] = (bf16)x.x; Asb[o + 1] = (bf16)x.y;
      Asb[o + 2] = (bf16)x.z; Asb[o + 3] = (bf16)x.w;
    }
    // stage B^T: [n][k]; global reads contiguous in n (coalesced b128)
#pragma unroll
    for (int i = 0; i < 4; ++i) {
      int q4 = t + i * 256;
      int k = q4 >> 5, n4 = q4 & 31;         // 32 float4 per k-row
      float4 w = Wv[(size_t)(kbase + k) * (NN / 4) + (n0 >> 2) + n4];
      int n = n4 * 4;
      Bsb[(n + 0) * BST + k] = (bf16)w.x;
      Bsb[(n + 1) * BST + k] = (bf16)w.y;
      Bsb[(n + 2) * BST + k] = (bf16)w.z;
      Bsb[(n + 3) * BST + k] = (bf16)w.w;
    }
    __syncthreads();

    v16bf bfr[4];
#pragma unroll
    for (int j = 0; j < 4; ++j)
      bfr[j] = frag_b_transposed(&Bsb[(wn * 64 + j * 16) * BST], BST, lane);
#pragma unroll
    for (int i = 0; i < 2; ++i) {
      v16bf afr = frag_a_rowmajor(&Asb[(wm * 32 + i * 16) * AST], AST, lane);
#pragma unroll
      for (int j = 0; j < 4; ++j)
        acc[i][j].v = __builtin_amdgcn_wmma_f32_16x16x32_bf16(
            false, afr, false, bfr[j], (short)0, acc[i][j].v, false, false);
    }
    __syncthreads();
  }

  const int half = lane >> 4;
  const int cl   = lane & 15;

  if (do_norm) {                                   // fused RMS over the head dim
    if (t < BM) rowstat[t] = 0.f;
    __syncthreads();
#pragma unroll
    for (int i = 0; i < 2; ++i)
#pragma unroll
      for (int j = 0; j < 4; ++j)
#pragma unroll
        for (int e = 0; e < 8; ++e) {
          int r = wm * 32 + i * 16 + e + half * 8;
          float v = acc[i][j].f[e];
          atomicAdd(&rowstat[r], v * v);           // ds_add_f32
        }
    __syncthreads();
    if (t < BM) rowstat[t] = rsqrtf(rowstat[t] * (1.0f / DD));
    __syncthreads();
  }

#pragma unroll
  for (int i = 0; i < 2; ++i)
#pragma unroll
    for (int j = 0; j < 4; ++j)
#pragma unroll
      for (int e = 0; e < 8; ++e) {
        int r = wm * 32 + i * 16 + e + half * 8;
        int d = wn * 64 + j * 16 + cl;
        float v = acc[i][j].f[e];
        if (do_norm) v *= rowstat[r];
        out[((size_t)head * MM + (m0 + r)) * DD + d] = (bf16)v;   // [H,M,D] bf16
      }
}

// ---------------- Flash attention over KV = cache ++ fresh ----------------
#define BQ  64
#define SKV 64
#define QST (DD + 4)    // 132
#define KST (DD + 4)    // 132  (K chunk [kv][d])
#define VST (SKV + 2)   // 66   (V chunk transposed [d][kv]); 128*66 == 64*132
#define SST (SKV + 4)   // 68

__global__ __launch_bounds__(256)
void attention_kernel(const bf16* __restrict__ qb, const bf16* __restrict__ kb,
                      const bf16* __restrict__ vb,
                      const float* __restrict__ cacheK, const float* __restrict__ cacheV,
                      float* __restrict__ out)
{
  __shared__ bf16  qs[BQ * QST];       // q tile [q][d]
  __shared__ bf16  KVs[SKV * KST];     // K chunk [kv][d]; aliased as Vt[d][kv]
  __shared__ float Sf[BQ * SST];       // scores
  __shared__ bf16  Pb[BQ * SST];       // probabilities (bf16)
  __shared__ float rowm[BQ], rowl[BQ], alpha[BQ];
  __shared__ float pm[256], ps[256];   // partial max / partial sum (4 threads/row)

  const int t    = threadIdx.x;
  const int lane = t & 31;
  const int wave = t >> 5;
  const int wm = wave & 3;             // row tile: wm*16
  const int wn = wave >> 2;            // S cols: wn*32 ; O cols: wn*64
  const int h  = blockIdx.y;
  const int q0 = blockIdx.x * BQ;
  const int half = lane >> 4;
  const int cl   = lane & 15;
  const int sr   = t >> 2;             // softmax row (4 threads per row)
  const int sq   = t & 3;              // softmax quarter

  const float4*  cKv = reinterpret_cast<const float4*>(cacheK);
  const float4*  cVv = reinterpret_cast<const float4*>(cacheV);
  const ushort4* kb4 = reinterpret_cast<const ushort4*>(kb);
  const ushort4* vb4 = reinterpret_cast<const ushort4*>(vb);
  const ushort4* qb4 = reinterpret_cast<const ushort4*>(qb);

  // stage normalized q tile (bf16 b64 copies)
#pragma unroll
  for (int i = 0; i < 8; ++i) {
    int q4 = t + i * 256;
    int r = q4 >> 5, d4 = q4 & 31;
    ushort4 raw = qb4[((size_t)h * MM + (q0 + r)) * (DD / 4) + d4];
    *reinterpret_cast<ushort4*>(&qs[r * QST + d4 * 4]) = raw;
  }
  if (t < BQ) { rowm[t] = -1e30f; rowl[t] = 0.f; }

  Acc oacc[4];
#pragma unroll
  for (int j = 0; j < 4; ++j)
#pragma unroll
    for (int e = 0; e < 8; ++e) oacc[j].f[e] = 0.f;

  for (int c0 = 0; c0 < KVLEN; c0 += SKV) {
    const bool incache = (c0 < PP);              // chunk-uniform (PP % SKV == 0)
    __syncthreads();                             // KVs/Pb free from previous chunk

    // ---- stage K chunk [kv][d] ----
    if (incache) {
#pragma unroll
      for (int i = 0; i < 8; ++i) {
        int q4 = t + i * 256;
        int kv = q4 >> 5, d4 = q4 & 31;
        float4 x = cKv[((size_t)h * PP + (c0 + kv)) * (DD / 4) + d4];
        int o = kv * KST + d4 * 4;
        KVs[o + 0] = (bf16)x.x; KVs[o + 1] = (bf16)x.y;
        KVs[o + 2] = (bf16)x.z; KVs[o + 3] = (bf16)x.w;
      }
    } else {
#pragma unroll
      for (int i = 0; i < 8; ++i) {
        int q4 = t + i * 256;
        int kv = q4 >> 5, d4 = q4 & 31;
        ushort4 raw = kb4[((size_t)h * MM + (c0 - PP + kv)) * (DD / 4) + d4];
        *reinterpret_cast<ushort4*>(&KVs[kv * KST + d4 * 4]) = raw;
      }
    }
    __syncthreads();

    // ---- S = q @ K^T (64x64), 4 k-steps over D ----
    Acc sacc[2];
#pragma unroll
    for (int j = 0; j < 2; ++j)
#pragma unroll
      for (int e = 0; e < 8; ++e) sacc[j].f[e] = 0.f;
#pragma unroll
    for (int ks = 0; ks < DD / 32; ++ks) {
      v16bf afr = frag_a_rowmajor(&qs[(wm * 16) * QST + ks * 32], QST, lane);
#pragma unroll
      for (int j = 0; j < 2; ++j) {
        v16bf bfr = frag_b_transposed(&KVs[(wn * 32 + j * 16) * KST + ks * 32], KST, lane);
        sacc[j].v = __builtin_amdgcn_wmma_f32_16x16x32_bf16(
            false, afr, false, bfr, (short)0, sacc[j].v, false, false);
      }
    }
#pragma unroll
    for (int j = 0; j < 2; ++j)
#pragma unroll
      for (int e = 0; e < 8; ++e) {
        int r = wm * 16 + e + half * 8;
        int c = wn * 32 + j * 16 + cl;
        Sf[r * SST + c] = sacc[j].f[e];
      }
    __syncthreads();

    // ---- stage V chunk transposed into Vt[d][kv] (aliases KVs) ----
    bf16* Vt = KVs;
    if (incache) {
#pragma unroll
      for (int i = 0; i < 8; ++i) {
        int q4 = t + i * 256;
        int kv = q4 >> 5, d4 = q4 & 31;
        float4 x = cVv[((size_t)h * PP + (c0 + kv)) * (DD / 4) + d4];
        Vt[(d4 * 4 + 0) * VST + kv] = (bf16)x.x;
        Vt[(d4 * 4 + 1) * VST + kv] = (bf16)x.y;
        Vt[(d4 * 4 + 2) * VST + kv] = (bf16)x.z;
        Vt[(d4 * 4 + 3) * VST + kv] = (bf16)x.w;
      }
    } else {
#pragma unroll
      for (int i = 0; i < 8; ++i) {
        int q4 = t + i * 256;
        int kv = q4 >> 5, d4 = q4 & 31;
        ushort4 raw = vb4[((size_t)h * MM + (c0 - PP + kv)) * (DD / 4) + d4];
        Vt[(d4 * 4 + 0) * VST + kv] = us2bf(raw.x);
        Vt[(d4 * 4 + 1) * VST + kv] = us2bf(raw.y);
        Vt[(d4 * 4 + 2) * VST + kv] = us2bf(raw.z);
        Vt[(d4 * 4 + 3) * VST + kv] = us2bf(raw.w);
      }
    }

    // ---- online softmax, 4 threads per row ----
    {
      float mc = -1e30f;
#pragma unroll
      for (int c = 0; c < 16; ++c) mc = fmaxf(mc, Sf[sr * SST + sq * 16 + c]);
      pm[t] = mc;
    }
    __syncthreads();
    if (t < BQ) {
      int r = t;
      float mnew = fmaxf(fmaxf(pm[4 * r], pm[4 * r + 1]),
                         fmaxf(pm[4 * r + 2], pm[4 * r + 3]));
      mnew = fmaxf(rowm[r], mnew);
      alpha[r] = __expf(rowm[r] - mnew);
      rowm[r] = mnew;
    }
    __syncthreads();
    {
      float m = rowm[sr], s = 0.f;
#pragma unroll
      for (int c = 0; c < 16; ++c) {
        float p = __expf(Sf[sr * SST + sq * 16 + c] - m);
        Pb[sr * SST + sq * 16 + c] = (bf16)p;
        s += p;
      }
      ps[t] = s;
    }
    __syncthreads();
    if (t < BQ) {
      int r = t;
      rowl[r] = rowl[r] * alpha[r] + (ps[4 * r] + ps[4 * r + 1]) +
                                     (ps[4 * r + 2] + ps[4 * r + 3]);
    }

    // ---- rescale O, then O += P @ V ----
    float alv[8];
#pragma unroll
    for (int e = 0; e < 8; ++e) alv[e] = alpha[wm * 16 + e + half * 8];
#pragma unroll
    for (int j = 0; j < 4; ++j)
#pragma unroll
      for (int e = 0; e < 8; ++e) oacc[j].f[e] *= alv[e];
#pragma unroll
    for (int ks = 0; ks < SKV / 32; ++ks) {
      v16bf afr = frag_a_rowmajor(&Pb[(wm * 16) * SST + ks * 32], SST, lane);
#pragma unroll
      for (int j = 0; j < 4; ++j) {
        v16bf bfr = frag_b_transposed(&Vt[(wn * 64 + j * 16) * VST + ks * 32], VST, lane);
        oacc[j].v = __builtin_amdgcn_wmma_f32_16x16x32_bf16(
            false, afr, false, bfr, (short)0, oacc[j].v, false, false);
      }
    }
  }

  __syncthreads();
  float lv[8];
#pragma unroll
  for (int e = 0; e < 8; ++e) lv[e] = rowl[wm * 16 + e + half * 8];
#pragma unroll
  for (int j = 0; j < 4; ++j)
#pragma unroll
    for (int e = 0; e < 8; ++e) {
      int r = wm * 16 + e + half * 8;
      int d = wn * 64 + j * 16 + cl;
      out[(size_t)(q0 + r) * NN + h * DD + d] = oacc[j].f[e] / lv[e];
    }
}

extern "C" void kernel_launch(void* const* d_in, const int* in_sizes, int n_in,
                              void* d_out, int out_size, void* d_ws, size_t ws_size,
                              hipStream_t stream) {
  (void)in_sizes; (void)n_in; (void)out_size; (void)ws_size;
  const float* X  = (const float*)d_in[0];
  const float* Wq = (const float*)d_in[1];
  const float* Wk = (const float*)d_in[2];
  const float* Wv = (const float*)d_in[3];
  const float* cK = (const float*)d_in[4];
  const float* cV = (const float*)d_in[5];
  float* out = (float*)d_out;

  const size_t elems = (size_t)HH * MM * DD;       // 4.19M per tensor
  bf16* qb = (bf16*)d_ws;
  bf16* kb = qb + elems;
  bf16* vb = kb + elems;

  dim3 gg(NN / BN, MM / BM);
  qkv_proj_kernel<<<gg, 256, 0, stream>>>(X, Wq, qb, 1);
  qkv_proj_kernel<<<gg, 256, 0, stream>>>(X, Wk, kb, 1);
  qkv_proj_kernel<<<gg, 256, 0, stream>>>(X, Wv, vb, 0);

  dim3 ga(MM / BQ, HH);
  attention_kernel<<<ga, 256, 0, stream>>>(qb, kb, vb, cK, cV, out);
}
